// PointNetSetAbstractionMSG_28638841929936
// MI455X (gfx1250) — matrix operand order repaired and checked by reading, not compiled, not run on hardware
//
#include <hip/hip_runtime.h>
#include <hip/hip_bf16.h>

#define Bn 8
#define Nn 16384
#define Sn 1024
#define BN_EPS 1e-05f

typedef __attribute__((ext_vector_type(16))) __bf16 v16bf;
typedef __attribute__((ext_vector_type(8)))  float  v8f;

union FragBF {
    v16bf v;
    unsigned u[8];
    uint4 q[2];
};

__device__ __forceinline__ float bf2f(unsigned short b) {
    return __builtin_bit_cast(float, ((unsigned)b) << 16);
}
__device__ __forceinline__ unsigned short f2bf(float f) {
    unsigned u = __builtin_bit_cast(unsigned, f);
    u += 0x7fffu + ((u >> 16) & 1u);   // round-to-nearest-even
    return (unsigned short)(u >> 16);
}

// ---------------------------------------------------------------------------
// 1) Farthest point sampling: one block per batch, 1024 sequential steps.
// ---------------------------------------------------------------------------
__global__ __launch_bounds__(256) void fps_kernel(const float* __restrict__ xyz,
                                                  float* __restrict__ dist,
                                                  int* __restrict__ cinds,
                                                  float* __restrict__ cent,
                                                  float* __restrict__ out_cent) {
    int b = blockIdx.x;
    const float* xs = xyz + (size_t)b * 3 * Nn;
    const float* ys = xs + Nn;
    const float* zs = xs + 2 * Nn;
    float* db = dist + (size_t)b * Nn;

    __shared__ float rv[256];
    __shared__ int   ri[256];
    __shared__ int   s_far;
    __shared__ float s_c[3];

    int tid = threadIdx.x;
    for (int n = tid; n < Nn; n += 256) db[n] = 3.4e38f;
    if (tid == 0) s_far = 0;
    __syncthreads();

    for (int it = 0; it < Sn; ++it) {
        if (tid == 0) {
            int far = s_far;
            cinds[b * Sn + it] = far;
            float cx = xs[far], cy = ys[far], cz = zs[far];
            s_c[0] = cx; s_c[1] = cy; s_c[2] = cz;
            size_t cb = ((size_t)b * Sn + it) * 3;
            cent[cb + 0] = cx; cent[cb + 1] = cy; cent[cb + 2] = cz;
            out_cent[(size_t)(b * 3 + 0) * Sn + it] = cx;
            out_cent[(size_t)(b * 3 + 1) * Sn + it] = cy;
            out_cent[(size_t)(b * 3 + 2) * Sn + it] = cz;
        }
        __syncthreads();
        float cx = s_c[0], cy = s_c[1], cz = s_c[2];
        float bestv = -1.0f;
        int   besti = 0;
        for (int n = tid; n < Nn; n += 256) {
            float dx = xs[n] - cx, dy = ys[n] - cy, dz = zs[n] - cz;
            float d = dx * dx + dy * dy + dz * dz;
            float nd = fminf(db[n], d);
            db[n] = nd;
            if (nd > bestv) { bestv = nd; besti = n; }
        }
        rv[tid] = bestv; ri[tid] = besti;
        __syncthreads();
        for (int s = 128; s > 0; s >>= 1) {
            if (tid < s) {
                if (rv[tid + s] > rv[tid] ||
                    (rv[tid + s] == rv[tid] && ri[tid + s] < ri[tid])) {
                    rv[tid] = rv[tid + s];
                    ri[tid] = ri[tid + s];
                }
            }
            __syncthreads();
        }
        if (tid == 0) s_far = ri[0];
        __syncthreads();
    }
}

// ---------------------------------------------------------------------------
// 2) Ball query: one wave32 per centroid; first-k in-ball indices in index
//    order (matches the reference's top_k(-idx)), pad with the first index.
// ---------------------------------------------------------------------------
__global__ __launch_bounds__(256) void query_ball_kernel(const float* __restrict__ xyz,
                                                         const float* __restrict__ cent,
                                                         int* __restrict__ grp,
                                                         float r2, int K) {
    int gw = (blockIdx.x * blockDim.x + threadIdx.x) >> 5;   // centroid id
    int lane = threadIdx.x & 31;
    if (gw >= Bn * Sn) return;                               // wave-uniform
    int b = gw / Sn;
    const float* xs = xyz + (size_t)b * 3 * Nn;
    const float* ys = xs + Nn;
    const float* zs = xs + 2 * Nn;
    float cx = cent[(size_t)gw * 3 + 0];
    float cy = cent[(size_t)gw * 3 + 1];
    float cz = cent[(size_t)gw * 3 + 2];
    int* g = grp + (size_t)gw * K;

    int cnt = 0;
    for (int base = 0; base < Nn && cnt < K; base += 32) {
        int n = base + lane;
        float dx = xs[n] - cx, dy = ys[n] - cy, dz = zs[n] - cz;
        float d = dx * dx + dy * dy + dz * dz;
        bool in = (d <= r2);
        unsigned mask = (unsigned)__ballot(in);
        int pos = cnt + __popc(mask & ((1u << lane) - 1u));
        if (in && pos < K) g[pos] = n;
        cnt += __popc(mask);
    }
    int cc = cnt < K ? cnt : K;
    int first = (cc > 0) ? g[0] : (Nn - 1);   // JAX gather clamps idx==N
    for (int j = lane; j < K; j += 32)
        if (j >= cc) g[j] = first;
}

// ---------------------------------------------------------------------------
// 3) Build layer-0 activations (bf16, 6 real channels zero-padded to 32).
// ---------------------------------------------------------------------------
__global__ __launch_bounds__(256) void build_x_kernel(const float* __restrict__ xyz,
                                                      const float* __restrict__ feat,
                                                      const float* __restrict__ cent,
                                                      const int* __restrict__ grp,
                                                      unsigned short* __restrict__ X,
                                                      int K) {
    size_t m = (size_t)blockIdx.x * blockDim.x + threadIdx.x;
    size_t total = (size_t)Bn * Sn * K;
    if (m >= total) return;
    size_t bs = m / K;
    int b = (int)(bs / Sn);
    int idx = grp[m];
    if (idx > Nn - 1) idx = Nn - 1;
    const float* xs = xyz + (size_t)b * 3 * Nn;
    const float* fs = feat + (size_t)b * 3 * Nn;
    float cx = cent[bs * 3 + 0], cy = cent[bs * 3 + 1], cz = cent[bs * 3 + 2];
    float vals[6] = { fs[idx], fs[Nn + idx], fs[2 * Nn + idx],
                      xs[idx] - cx, xs[Nn + idx] - cy, xs[2 * Nn + idx] - cz };
    unsigned short* row = X + m * 32;
#pragma unroll
    for (int c = 0; c < 6; ++c) row[c] = f2bf(vals[c]);
#pragma unroll
    for (int c = 6; c < 32; ++c) row[c] = 0;
}

// ---------------------------------------------------------------------------
// 4) Pack fp32 weights -> bf16, cin padded to a multiple of 32.
// ---------------------------------------------------------------------------
__global__ __launch_bounds__(256) void conv_w_kernel(const float* __restrict__ W,
                                                     unsigned short* __restrict__ Wb,
                                                     int cout, int cin, int cinpad) {
    int t = blockIdx.x * blockDim.x + threadIdx.x;
    if (t >= cout * cinpad) return;
    int o = t / cinpad, c = t % cinpad;
    Wb[t] = f2bf((c < cin) ? W[o * cin + c] : 0.0f);
}

// ---------------------------------------------------------------------------
// 5) WMMA GEMM: Y = normReLU(X)*W^T + b.
//    Each wave computes a 16x64 strip: one A fragment feeds 4 column tiles
//    (NT=4), cutting A traffic 4x. CINPAD/APPLY are compile-time so the
//    K-loop fully unrolls into CINPAD/32 * 4 back-to-back v_wmma ops.
//    Fragment layouts per CDNA5 ISA 7.12.2 (16-bit, wave32).
// ---------------------------------------------------------------------------
template <int CINPAD, int APPLY>
__global__ __launch_bounds__(256) void gemm_wmma_nt4(const unsigned short* __restrict__ X,
                                                     const unsigned short* __restrict__ Wb,
                                                     const float* __restrict__ bias,
                                                     const float* __restrict__ scaleA,
                                                     const float* __restrict__ shiftA,
                                                     unsigned short* __restrict__ Y,
                                                     int M, int cout) {
    const int wave = threadIdx.x >> 5;
    const int lane = threadIdx.x & 31;
    const int groups = cout >> 6;                 // 64 output cols per wave
    const int tile = blockIdx.x * 8 + wave;
    const int mtile = tile / groups;
    const int gidx = tile % groups;
    if (mtile >= (M >> 4)) return;                // wave-uniform guard

    const int half = lane >> 4;
    const int r = lane & 15;
    const int m = (mtile << 4) + r;               // A row held by this lane
    const int colbase = gidx << 6;
    const unsigned short* arow = X + (size_t)m * CINPAD;

    v8f acc[4] = {};
#pragma unroll
    for (int k0 = 0; k0 < CINPAD; k0 += 32) {
        FragBF a;
        // A: lanes 0-15 hold K={0..7,16..23}; lanes 16-31 K={8..15,24..31}
        a.q[0] = *(const uint4*)(arow + k0 + half * 8);
        a.q[1] = *(const uint4*)(arow + k0 + 16 + half * 8);
        if (APPLY) {                              // fused prev-layer BN + ReLU
#pragma unroll
            for (int p = 0; p < 8; ++p) {
                int kb = k0 + ((p < 4) ? (half * 8 + 2 * p)
                                       : (16 + half * 8 + 2 * (p - 4)));
                unsigned w = a.u[p];
                float lo = fmaxf(bf2f((unsigned short)(w & 0xffffu)) * scaleA[kb] + shiftA[kb], 0.0f);
                float hi = fmaxf(bf2f((unsigned short)(w >> 16)) * scaleA[kb + 1] + shiftA[kb + 1], 0.0f);
                a.u[p] = (unsigned)f2bf(lo) | ((unsigned)f2bf(hi) << 16);
            }
        }
#pragma unroll
        for (int nt = 0; nt < 4; ++nt) {
            // B: lanes 0-15 hold K={0..7,16..23}; lanes 16-31 K={8..15,24..31} of column n
            const unsigned short* brow =
                Wb + (size_t)(colbase + (nt << 4) + r) * CINPAD + k0 + half * 8;
            FragBF bfr;
            bfr.q[0] = ((const uint4*)brow)[0];
            bfr.q[1] = ((const uint4*)(brow + 16))[0];
            acc[nt] = __builtin_amdgcn_wmma_f32_16x16x32_bf16(
                false, a.v, false, bfr.v, (short)0, acc[nt], false, false);
        }
    }

    const int mbase = (mtile << 4) + (half << 3); // C/D: VGPR j -> row j+8*half
#pragma unroll
    for (int nt = 0; nt < 4; ++nt) {
        const int n = colbase + (nt << 4) + r;
        const float bn = bias[n];
#pragma unroll
        for (int j = 0; j < 8; ++j) {
            Y[(size_t)(mbase + j) * cout + n] = f2bf(acc[nt][j] + bn);
        }
    }
}

// ---------------------------------------------------------------------------
// 6) BN statistics: per-channel sum & sumsq (fp32), atomics across blocks.
// ---------------------------------------------------------------------------
__global__ __launch_bounds__(256) void bn_reduce(const unsigned short* __restrict__ Y,
                                                 float* __restrict__ stats,
                                                 int M, int cout) {
    __shared__ float ss[256], sq[256];
    int tid = threadIdx.x;
    int tpc = 256 / cout;                  // cout in {64,128,256}
    int c = tid % cout;
    int ridx = tid / cout;
    const int ROWS = 2048;
    int r0 = blockIdx.x * ROWS;
    int r1 = min(M, r0 + ROWS);
    float s = 0.0f, s2 = 0.0f;
    for (int r = r0 + ridx; r < r1; r += tpc) {
        float v = bf2f(Y[(size_t)r * cout + c]);
        s += v;
        s2 += v * v;
    }
    ss[tid] = s; sq[tid] = s2;
    __syncthreads();
    if (ridx == 0) {
        for (int i = 1; i < tpc; ++i) { s += ss[tid + i * cout]; s2 += sq[tid + i * cout]; }
        atomicAdd(&stats[c], s);
        atomicAdd(&stats[256 + c], s2);
    }
}

__global__ void zero_kernel(float* p, int n) {
    int t = blockIdx.x * blockDim.x + threadIdx.x;
    if (t < n) p[t] = 0.0f;
}

__global__ void bn_scale_kernel(const float* __restrict__ stats,
                                const float* __restrict__ gamma,
                                const float* __restrict__ beta,
                                float* __restrict__ scale,
                                float* __restrict__ shift,
                                int M, int cout) {
    int c = threadIdx.x;
    if (c >= cout) return;
    float inv = 1.0f / (float)M;
    float mean = stats[c] * inv;
    float var = stats[256 + c] * inv - mean * mean;
    float sc = gamma[c] * rsqrtf(var + BN_EPS);
    scale[c] = sc;
    shift[c] = beta[c] - mean * sc;
}

// ---------------------------------------------------------------------------
// 7) Final BN + ReLU + max over k, write (B, 512, S) feature output section.
// ---------------------------------------------------------------------------
__global__ __launch_bounds__(256) void pool_kernel(const unsigned short* __restrict__ Y,
                                                   const float* __restrict__ scale,
                                                   const float* __restrict__ shift,
                                                   float* __restrict__ out,
                                                   int K, int branch) {
    int t = blockIdx.x * blockDim.x + threadIdx.x;
    if (t >= Bn * Sn * 256) return;
    int c = t & 255;
    int bs = t >> 8;
    int b = bs / Sn, s = bs % Sn;
    float sc = scale[c], sh = shift[c];
    const unsigned short* base = Y + ((size_t)bs * K) * 256 + c;
    float mx = 0.0f;                        // max of ReLU'd values is >= 0
    for (int kk = 0; kk < K; ++kk) {
        float v = fmaxf(bf2f(base[(size_t)kk * 256]) * sc + sh, 0.0f);
        mx = fmaxf(mx, v);
    }
    out[24576 + ((size_t)(b * 512 + branch * 256 + c)) * Sn + s] = mx;
}

// ---------------------------------------------------------------------------
extern "C" void kernel_launch(void* const* d_in, const int* in_sizes, int n_in,
                              void* d_out, int out_size, void* d_ws, size_t ws_size,
                              hipStream_t stream) {
    const float* xyz = (const float*)d_in[0];      // (B,3,N)
    const float* feat = (const float*)d_in[1];     // (B,3,N)
    float* out = (float*)d_out;

    // Layer shapes
    const int cin[6]    = {6, 64, 64, 128, 128, 128};
    const int cinpad[6] = {32, 64, 64, 128, 128, 128};
    const int coutA[6]  = {64, 64, 128, 128, 128, 256};

    // Workspace layout
    char* base = (char*)d_ws;
    size_t o = 0;
    auto alloc = [&](size_t bytes) { size_t r = o; o += (bytes + 255) & ~(size_t)255; return r; };
    float* dist  = (float*)(base + alloc((size_t)Bn * Nn * 4));
    float* cent  = (float*)(base + alloc((size_t)Bn * Sn * 3 * 4));
    int*   cinds = (int*)  (base + alloc((size_t)Bn * Sn * 4));
    int*   grp0  = (int*)  (base + alloc((size_t)Bn * Sn * 16 * 4));
    int*   grp1  = (int*)  (base + alloc((size_t)Bn * Sn * 32 * 4));
    unsigned short* Wb[6];
    for (int l = 0; l < 6; ++l)
        Wb[l] = (unsigned short*)(base + alloc((size_t)coutA[l] * cinpad[l] * 2));
    float* stats = (float*)(base + alloc(512 * 4));
    float* scl   = (float*)(base + alloc(256 * 4));
    float* shf   = (float*)(base + alloc(256 * 4));
    const size_t maxM = (size_t)Bn * Sn * 32;
    unsigned short* bufA = (unsigned short*)(base + alloc(maxM * 256 * 2));
    unsigned short* bufB = (unsigned short*)(base + alloc(maxM * 256 * 2));

    // 1) FPS + centroid gather (+ centroid output section)
    fps_kernel<<<Bn, 256, 0, stream>>>(xyz, dist, cinds, cent, out);

    // 2) Ball queries
    const float radii2[2] = {0.1f * 0.1f, 0.2f * 0.2f};
    const int   Ks[2] = {16, 32};
    int* grps[2] = {grp0, grp1};
    for (int r = 0; r < 2; ++r)
        query_ball_kernel<<<(Bn * Sn) / 8, 256, 0, stream>>>(xyz, cent, grps[r],
                                                             radii2[r], Ks[r]);

    // 3) Pack weights to bf16 (padded)
    for (int l = 0; l < 6; ++l) {
        const float* W = (const float*)d_in[2 + 4 * l];
        int total = coutA[l] * cinpad[l];
        conv_w_kernel<<<(total + 255) / 256, 256, 0, stream>>>(W, Wb[l], coutA[l],
                                                               cin[l], cinpad[l]);
    }

    // 4) Two branches: build X0, then 6x (GEMM -> stats -> scale/shift), pool
    for (int r = 0; r < 2; ++r) {
        int K = Ks[r];
        int M = Bn * Sn * K;
        build_x_kernel<<<(M + 255) / 256, 256, 0, stream>>>(xyz, feat, cent,
                                                            grps[r], bufA, K);
        unsigned short* X = bufA;
        unsigned short* Y = bufB;
        for (int l = 0; l < 6; ++l) {
            const float* bia   = (const float*)d_in[3 + 4 * l];
            const float* gamma = (const float*)d_in[4 + 4 * l];
            const float* beta  = (const float*)d_in[5 + 4 * l];
            int blocks = ((M >> 4) * (coutA[l] >> 6)) / 8;   // waves/8
            switch (cinpad[l] * 10 + (l > 0 ? 1 : 0)) {
                case 320:  // layer 0: cinpad=32, no norm
                    gemm_wmma_nt4<32, 0><<<blocks, 256, 0, stream>>>(
                        X, Wb[l], bia, scl, shf, Y, M, coutA[l]);
                    break;
                case 641:  // layers 1,2: cinpad=64, norm
                    gemm_wmma_nt4<64, 1><<<blocks, 256, 0, stream>>>(
                        X, Wb[l], bia, scl, shf, Y, M, coutA[l]);
                    break;
                default:   // layers 3,4,5: cinpad=128, norm
                    gemm_wmma_nt4<128, 1><<<blocks, 256, 0, stream>>>(
                        X, Wb[l], bia, scl, shf, Y, M, coutA[l]);
                    break;
            }
            zero_kernel<<<2, 256, 0, stream>>>(stats, 512);
            bn_reduce<<<(M + 2047) / 2048, 256, 0, stream>>>(Y, stats, M, coutA[l]);
            bn_scale_kernel<<<1, 256, 0, stream>>>(stats, gamma, beta, scl, shf,
                                                   M, coutA[l]);
            unsigned short* t = X; X = Y; Y = t;   // ping-pong
        }
        // X now holds the raw layer-5 output; apply final BN+ReLU in the pool
        int total = Bn * Sn * 256;
        pool_kernel<<<(total + 255) / 256, 256, 0, stream>>>(X, scl, shf, out, K, r);
    }
}
